// MultiHeadAttention_86517821216540
// MI455X (gfx1250) — matrix operand order repaired and checked by reading, not compiled, not run on hardware
//
#include <hip/hip_runtime.h>
#include <math.h>

// MultiHeadAttention forward for MI455X (gfx1250, wave32, WMMA).
// Compute-bound (~137 GFLOP vs ~200MB traffic @ 23.3TB/s) -> bf16 WMMA path.
// GEMM tiles stream global->LDS via CDNA5 async copies (ASYNCcnt), double-
// buffered so the DMA overlaps the WMMA pipe (ASYNCcnt retires in-order, so
// "s_wait_asynccnt 4" after issuing the next tile's 4 copies guarantees the
// previous tile's 4 copies have completed).
#define BB 4
#define SS 2048
#define DD 1024
#define HH 16
#define HD 64
#define MM (BB * SS) // 8192

typedef __bf16 bf16;
typedef __attribute__((ext_vector_type(16))) __bf16 v16bf;
typedef __attribute__((ext_vector_type(8)))  __bf16 v8bf;
typedef __attribute__((ext_vector_type(8)))  float  v8f;

__device__ __forceinline__ v16bf cat8(v8bf lo, v8bf hi) {
  return __builtin_shufflevector(lo, hi, 0,1,2,3,4,5,6,7,8,9,10,11,12,13,14,15);
}

__device__ __forceinline__ v8f wmma_bf16(v16bf a, v16bf b, v8f c) {
  // D = A x B + C, 16x16x32, f32 accum
  return __builtin_amdgcn_wmma_f32_16x16x32_bf16(false, a, false, b, (short)0, c, false, false);
}

// CDNA5 one-hop global->LDS async copy, 16B per lane, tracked by ASYNCcnt.
__device__ __forceinline__ void async_ld_b128(const bf16* lds_dst, const bf16* gsrc) {
  unsigned int l = (unsigned int)(unsigned long long)lds_dst;
  asm volatile("global_load_async_to_lds_b128 %0, %1, off"
               :: "v"(l), "v"(gsrc) : "memory");
}
__device__ __forceinline__ void wait_async0() {
  asm volatile("s_wait_asynccnt 0" ::: "memory");
}
__device__ __forceinline__ void wait_async4() {
  // in-order retirement: <=4 outstanding => everything older than the last
  // 4 issued (i.e. the previous tile) has landed in LDS
  asm volatile("s_wait_asynccnt 4" ::: "memory");
}

// ---------------------------------------------------------------------------
// One-time conversions (negligible vs matrix math):
//   activations fp32 -> bf16 (row-major kept)
//   weights     fp32 -> bf16 AND transposed to [N][K] so GEMM B-tiles are
//               contiguous 16B chunks -> pure async copies, no LDS transpose.
// ---------------------------------------------------------------------------
__global__ void cvt_f32_bf16(const float* __restrict__ src, bf16* __restrict__ dst, int n) {
  int i = blockIdx.x * blockDim.x + threadIdx.x;
  if (i < n) dst[i] = (bf16)src[i];
}

__global__ void cvt_transpose_w(const float* __restrict__ W, bf16* __restrict__ Wt) {
  // W: [K=1024][N=1024] row-major -> Wt: [N][K]
  int idx = blockIdx.x * blockDim.x + threadIdx.x; // 1M elements
  int n = idx >> 10, k = idx & 1023;
  Wt[((size_t)n << 10) + k] = (bf16)W[((size_t)k << 10) + n];
}

// ---------------------------------------------------------------------------
// All-bf16 WMMA GEMM: C[M,1024] = A[M,1024] * Wt^T  (Wt stored [N][K])
// OUTMODE 0: bf16 head-split [B,H,S,64]      (Q, K projections)
// OUTMODE 2: bf16 head-split-T [B,H,64,S]    (V projection -> flash B-frags
//                                             become direct global loads)
// OUTMODE 1: fp32 flat [M,N] + bias          (output projection)
// 256 threads (8 waves), 128x128 tile, wave = 64x32 (4x2 accums), K-step 32.
// Double-buffered async global->LDS pipeline.
// ---------------------------------------------------------------------------
template <int OUTMODE>
__global__ __launch_bounds__(256)
void gemm_wmma(const bf16* __restrict__ A, const bf16* __restrict__ Bt,
               void* __restrict__ Out, const float* __restrict__ bias) {
  const int K = DD, N = DD;
  const int KT = K / 32; // 32 k-tiles
  __shared__ __attribute__((aligned(16))) bf16 As[2][128 * 40]; // [m][k], pad 32->40
  __shared__ __attribute__((aligned(16))) bf16 Bs[2][128 * 40]; // [n][k], pad 32->40

  const int tid  = threadIdx.x;
  const int lane = tid & 31;
  const int wave = tid >> 5;
  const int half = lane >> 4;
  const int ln   = lane & 15;
  const int wm   = wave & 1;   // 2 M sub-blocks of 64
  const int wn   = wave >> 1;  // 4 N sub-blocks of 32
  const int bm   = blockIdx.y * 128;
  const int bn   = blockIdx.x * 128;

  v8f acc[4][2];
#pragma unroll
  for (int mt = 0; mt < 4; ++mt)
#pragma unroll
    for (int nt = 0; nt < 2; ++nt)
#pragma unroll
      for (int e = 0; e < 8; ++e) acc[mt][nt][e] = 0.f;

  // issue one 128x32 bf16 tile pair as 2x2 async 16B chunk copies per thread
  auto issue_tile = [&](int k0, int buf) {
#pragma unroll
    for (int i = 0; i < 2; ++i) {
      const int c  = tid + i * 256;   // 0..511
      const int r  = c >> 2;          // row (m or n)
      const int ck = (c & 3) * 8;     // k sub-chunk (halves)
      async_ld_b128(&As[buf][r * 40 + ck], A  + (size_t)(bm + r) * K + k0 + ck);
      async_ld_b128(&Bs[buf][r * 40 + ck], Bt + (size_t)(bn + r) * K + k0 + ck);
    }
  };

  issue_tile(0, 0); // prologue

  for (int kt = 0; kt < KT; ++kt) {
    const int cur = kt & 1;
    if (kt + 1 < KT) {
      issue_tile((kt + 1) * 32, cur ^ 1); // stream next tile while computing
      wait_async4();                      // previous (current) tile landed
    } else {
      wait_async0();
    }
    __syncthreads();

    // ---- fragments (ISA 16-bit A 16x32 / B 32x16 layouts) ----
    v16bf af[4];
#pragma unroll
    for (int mt = 0; mt < 4; ++mt) {
      const int r = wm * 64 + mt * 16 + ln;
      v8bf lo = *(const v8bf*)&As[cur][r * 40 + 8 * half];
      v8bf hi = *(const v8bf*)&As[cur][r * 40 + 16 + 8 * half];
      af[mt] = cat8(lo, hi);
    }
    v16bf bfv[2];
#pragma unroll
    for (int nt = 0; nt < 2; ++nt) {
      const int c = wn * 32 + nt * 16 + ln;
      v8bf lo = *(const v8bf*)&Bs[cur][c * 40 + 8 * half];
      v8bf hi = *(const v8bf*)&Bs[cur][c * 40 + 16 + 8 * half];
      bfv[nt] = cat8(lo, hi);
    }
#pragma unroll
    for (int mt = 0; mt < 4; ++mt)
#pragma unroll
      for (int nt = 0; nt < 2; ++nt)
        acc[mt][nt] = wmma_bf16(af[mt], bfv[nt], acc[mt][nt]);
    __syncthreads(); // this buffer may be re-filled next iteration
  }

  // ---- writeback (C/D layout: lane holds M = 8*half + i, N = ln) ----
#pragma unroll
  for (int mt = 0; mt < 4; ++mt) {
#pragma unroll
    for (int nt = 0; nt < 2; ++nt) {
      const int n_g = bn + wn * 32 + nt * 16 + ln;
#pragma unroll
      for (int i = 0; i < 8; ++i) {
        const int m_g = bm + wm * 64 + mt * 16 + 8 * half + i;
        const float v = acc[mt][nt][i];
        const int b = m_g >> 11;   // / 2048
        const int s = m_g & 2047;
        const int h = n_g >> 6;
        const int d = n_g & 63;
        if (OUTMODE == 0) {
          ((bf16*)Out)[((((size_t)b * HH + h) * SS + s) << 6) + d] = (bf16)v;
        } else if (OUTMODE == 2) {
          ((bf16*)Out)[((((size_t)b * HH + h) * HD + d) << 11) + s] = (bf16)v;
        } else {
          ((float*)Out)[(size_t)m_g * N + n_g] = v + bias[n_g];
        }
      }
    }
  }
}

// ---------------------------------------------------------------------------
// Flash attention (causal) per (b*h, 64-row query block).
// 128 threads = 4 waves; wave w owns rows [qblk*64 + 16w, +16); waves are
// fully independent (no barriers): Q/K/V fragments all come straight from
// global in the right layout; LDS only relays P (C/D -> A-fragment).
// ---------------------------------------------------------------------------
__global__ __launch_bounds__(128)
void flash_attn(const bf16* __restrict__ Qh, const bf16* __restrict__ Kh,
                const bf16* __restrict__ Vt, bf16* __restrict__ Oattn) {
  __shared__ __attribute__((aligned(16))) bf16 Ps[4][16 * 40]; // per-wave P tile

  const int tid  = threadIdx.x;
  const int lane = tid & 31;
  const int wave = tid >> 5;
  const int half = lane >> 4;
  const int ln   = lane & 15;
  const int bh   = blockIdx.x;
  const int b    = bh >> 4;
  const int h    = bh & 15;
  const int qblk = blockIdx.y;
  const int q0   = qblk * 64 + wave * 16;

  const bf16* Qb = Qh + (size_t)bh * SS * HD; // [s][d]
  const bf16* Kb = Kh + (size_t)bh * SS * HD; // [s][d]
  const bf16* Vb = Vt + (size_t)bh * HD * SS; // [d][s]  (transposed by proj)

  // Q A-fragments for the 2 k-steps of the 64-deep QK^T reduction
  v16bf qf[2];
  {
    const int qr = q0 + ln;
#pragma unroll
    for (int ks = 0; ks < 2; ++ks) {
      v8bf lo = *(const v8bf*)&Qb[(size_t)qr * HD + 32 * ks + 8 * half];
      v8bf hi = *(const v8bf*)&Qb[(size_t)qr * HD + 32 * ks + 16 + 8 * half];
      qf[ks] = cat8(lo, hi);
    }
  }

  v8f o[4];
#pragma unroll
  for (int nt = 0; nt < 4; ++nt)
#pragma unroll
    for (int e = 0; e < 8; ++e) o[nt][e] = 0.f;
  float m_run[8], l_run[8];
#pragma unroll
  for (int i = 0; i < 8; ++i) { m_run[i] = -INFINITY; l_run[i] = 0.f; }

  const float scale = 0.125f;   // 1/sqrt(64)
  const int njb = qblk * 2 + 2; // causal: kv blocks covering cols <= qblk*64+63

  for (int j = 0; j < njb; ++j) {
    const int kv0 = j * 32;

    // prefetch next KV block into cache (global_prefetch_b8)
    if (j + 1 < njb) {
      const int nkv = kv0 + 32;
      __builtin_prefetch(&Kb[(size_t)(nkv + ln) * HD + 32 * half], 0, 1);
      __builtin_prefetch(&Vb[(size_t)(lane << 1) * SS + nkv], 0, 1);
    }

    // ---- logits S = Q K^T (16x32 tile, f32) ----
    v8f sa[2];
#pragma unroll
    for (int nt = 0; nt < 2; ++nt)
#pragma unroll
      for (int e = 0; e < 8; ++e) sa[nt][e] = 0.f;
#pragma unroll
    for (int nt = 0; nt < 2; ++nt) {
      const int col = kv0 + nt * 16 + ln;
#pragma unroll
      for (int ks = 0; ks < 2; ++ks) {
        const bf16* kp = &Kb[(size_t)col * HD + 32 * ks + 8 * half];
        v16bf kf = cat8(*(const v8bf*)kp, *(const v8bf*)(kp + 16));
        sa[nt] = wmma_bf16(qf[ks], kf, sa[nt]);
      }
    }

    // ---- scale + causal mask ----
#pragma unroll
    for (int nt = 0; nt < 2; ++nt) {
      const int col = kv0 + nt * 16 + ln;
#pragma unroll
      for (int i = 0; i < 8; ++i) {
        const int row = q0 + 8 * half + i;
        const float v = sa[nt][i] * scale;
        sa[nt][i] = (col <= row) ? v : -1e30f;
      }
    }

    // ---- online softmax (row stats across the 16-lane N group) ----
#pragma unroll
    for (int i = 0; i < 8; ++i) {
      float v = fmaxf(sa[0][i], sa[1][i]);
      v = fmaxf(v, __shfl_xor(v, 1, 32));
      v = fmaxf(v, __shfl_xor(v, 2, 32));
      v = fmaxf(v, __shfl_xor(v, 4, 32));
      v = fmaxf(v, __shfl_xor(v, 8, 32));
      const float mnew = fmaxf(m_run[i], v);
      const float corr = __expf(m_run[i] - mnew);
      m_run[i] = mnew;
      const float p0 = __expf(sa[0][i] - mnew);
      const float p1 = __expf(sa[1][i] - mnew);
      sa[0][i] = p0; sa[1][i] = p1;
      float rs = p0 + p1;
      rs += __shfl_xor(rs, 1, 32);
      rs += __shfl_xor(rs, 2, 32);
      rs += __shfl_xor(rs, 4, 32);
      rs += __shfl_xor(rs, 8, 32);
      l_run[i] = l_run[i] * corr + rs;
      o[0][i] *= corr; o[1][i] *= corr; o[2][i] *= corr; o[3][i] *= corr;
    }

    // ---- P (C/D layout) -> per-wave LDS -> A-fragment ----
    bf16* ps = &Ps[wave][0];
#pragma unroll
    for (int nt = 0; nt < 2; ++nt)
#pragma unroll
      for (int i = 0; i < 8; ++i)
        ps[(8 * half + i) * 40 + nt * 16 + ln] = (bf16)sa[nt][i];
    v16bf pf;
    {
      v8bf lo = *(const v8bf*)&ps[ln * 40 + 8 * half];
      v8bf hi = *(const v8bf*)&ps[ln * 40 + 16 + 8 * half];
      pf = cat8(lo, hi);
    }

    // ---- O += P x V (V^T global: column d contiguous over kv) ----
#pragma unroll
    for (int nt = 0; nt < 4; ++nt) {
      const int d = nt * 16 + ln;
      const bf16* vp = &Vb[(size_t)d * SS + kv0 + 8 * half];
      v16bf vf = cat8(*(const v8bf*)vp, *(const v8bf*)(vp + 16));
      o[nt] = wmma_bf16(pf, vf, o[nt]);
    }
  }

  // ---- epilogue: O / l, write concat-head layout [B,S,H*64] ----
#pragma unroll
  for (int nt = 0; nt < 4; ++nt) {
#pragma unroll
    for (int i = 0; i < 8; ++i) {
      const int row = q0 + 8 * half + i;
      const int d   = nt * 16 + ln;
      const float val = o[nt][i] / l_run[i];
      Oattn[((size_t)b * SS + row) * DD + h * HD + d] = (bf16)val;
    }
  }
}

// ---------------------------------------------------------------------------
// Launcher. Workspace layout (120 MiB needed):
//   [0,16M)    Xq bf16      [48M,64M)  Qh bf16 [B,H,S,64]   [96M,112M) Oattn
//   [16M,32M)  Xk bf16      [64M,80M)  Kh bf16 [B,H,S,64]   [112M..]   Wt bf16
//   [32M,48M)  Xv bf16      [80M,96M)  Vt bf16 [B,H,64,S]              (2M ea)
// ---------------------------------------------------------------------------
extern "C" void kernel_launch(void* const* d_in, const int* in_sizes, int n_in,
                              void* d_out, int out_size, void* d_ws, size_t ws_size,
                              hipStream_t stream) {
  const float* q  = (const float*)d_in[0];
  const float* k  = (const float*)d_in[1];
  const float* v  = (const float*)d_in[2];
  // d_in[3] = causal mask: handled analytically in flash_attn
  const float* Wq = (const float*)d_in[4];
  const float* Wk = (const float*)d_in[5];
  const float* Wv = (const float*)d_in[6];
  const float* Wo = (const float*)d_in[7];
  const float* bo = (const float*)d_in[8];

  char* ws = (char*)d_ws;
  bf16* Xq  = (bf16*)(ws);
  bf16* Xk  = (bf16*)(ws + ((size_t)16 << 20));
  bf16* Xv  = (bf16*)(ws + ((size_t)32 << 20));
  bf16* Qh  = (bf16*)(ws + ((size_t)48 << 20));
  bf16* Kh  = (bf16*)(ws + ((size_t)64 << 20));
  bf16* Vt  = (bf16*)(ws + ((size_t)80 << 20));
  bf16* Oa  = (bf16*)(ws + ((size_t)96 << 20));
  bf16* Wqt = (bf16*)(ws + ((size_t)112 << 20));
  bf16* Wkt = (bf16*)(ws + ((size_t)114 << 20));
  bf16* Wvt = (bf16*)(ws + ((size_t)116 << 20));
  bf16* Wot = (bf16*)(ws + ((size_t)118 << 20));

  const int xn = MM * DD;  // 8M activations
  cvt_f32_bf16<<<xn / 256, 256, 0, stream>>>(q, Xq, xn);
  cvt_f32_bf16<<<xn / 256, 256, 0, stream>>>(k, Xk, xn);
  cvt_f32_bf16<<<xn / 256, 256, 0, stream>>>(v, Xv, xn);
  const int wn = DD * DD;  // 1M per weight
  cvt_transpose_w<<<wn / 256, 256, 0, stream>>>(Wq, Wqt);
  cvt_transpose_w<<<wn / 256, 256, 0, stream>>>(Wk, Wkt);
  cvt_transpose_w<<<wn / 256, 256, 0, stream>>>(Wv, Wvt);
  cvt_transpose_w<<<wn / 256, 256, 0, stream>>>(Wo, Wot);

  dim3 gg(DD / 128, MM / 128); // 8 x 64 tiles
  gemm_wmma<0><<<gg, 256, 0, stream>>>(Xq, Wqt, (void*)Qh, nullptr);
  gemm_wmma<0><<<gg, 256, 0, stream>>>(Xk, Wkt, (void*)Kh, nullptr);
  gemm_wmma<2><<<gg, 256, 0, stream>>>(Xv, Wvt, (void*)Vt, nullptr);

  flash_attn<<<dim3(BB * HH, SS / 64), 128, 0, stream>>>(Qh, Kh, Vt, Oa);

  gemm_wmma<1><<<gg, 256, 0, stream>>>(Oa, Wot, d_out, bo);

  (void)in_sizes; (void)n_in; (void)out_size; (void)ws_size;
}
